// AttentionModule_26628797235756
// MI455X (gfx1250) — matrix operand order repaired and checked by reading, not compiled, not run on hardware
//
#include <hip/hip_runtime.h>
#include <hip/hip_bf16.h>
#include <math.h>

typedef __attribute__((ext_vector_type(16))) _Float16 v16h;
typedef __attribute__((ext_vector_type(8)))  _Float16 v8h;
typedef __attribute__((ext_vector_type(8)))  float    v8f;
typedef __attribute__((ext_vector_type(4)))  unsigned int u32x4;
typedef __attribute__((ext_vector_type(8)))  int      i32x8;
typedef __attribute__((ext_vector_type(4)))  int      i32x4;

#define BB  8
#define NN1 4096
#define NN2 4096
#define DXX 128
#define HHH 256
#define MROWS (BB*NN1)   // 32768 rows through each MLP

// ---------- WMMA tile loaders (wave32, contiguous k order, same in A and B) ----------

// A-matrix 16x32 f16, row-major source X (leading dim ld).
// lane: m = lane&15, half = lane>>4; element i <-> k = 16*half + i
__device__ __forceinline__ v16h load_a_tile(const _Float16* __restrict__ X,
                                            int ld, int row0, int k0) {
  int lane = threadIdx.x & 31;
  int l15 = lane & 15, lh = lane >> 4;
  const _Float16* p = X + (size_t)(row0 + l15) * ld + k0 + lh * 16;
  v8h lo = *(const v8h*)p;          // k = 16h .. 16h+7
  v8h hi = *(const v8h*)(p + 8);    // k = 16h+8 .. 16h+15
  return __builtin_shufflevector(lo, hi, 0,1,2,3,4,5,6,7,8,9,10,11,12,13,14,15);
}

// B-matrix 32x16 f16 from "B^T row-major" source T: T[n][k], leading dim ld.
// lane: n = lane&15, half = lane>>4; element i <-> k = 16*half + i
__device__ __forceinline__ v16h load_b_tile(const _Float16* __restrict__ T,
                                            int ld, int n0, int k0) {
  int lane = threadIdx.x & 31;
  int l15 = lane & 15, lh = lane >> 4;
  const _Float16* p = T + (size_t)(n0 + l15) * ld + k0 + lh * 16;
  return *(const v16h*)p;           // 32B contiguous
}

// ---------- Tensor Data Mover: 2D tile global->LDS with row padding ----------
// D# per CDNA5 ISA §8.3-8.5. data_size=1 (2 bytes). pad_enable=1.
// This toolchain exposes the 6-arg builtin: (g0, g1, g2, g3, g4, cpol).
__device__ __forceinline__ void tdm_load_2d(unsigned lds_byte_off,
                                            const void* gptr,
                                            unsigned tensor_d0, unsigned tensor_d1,
                                            unsigned tile_d0,   unsigned tile_d1,
                                            unsigned long long stride_d0,
                                            unsigned pad_int_code, unsigned pad_amt_code) {
  unsigned long long ga = (unsigned long long)gptr;
  u32x4 g0;
  g0[0] = 1u;                                           // count=1 (valid user D#)
  g0[1] = lds_byte_off;                                 // lds_addr
  g0[2] = (unsigned)ga;                                 // global_addr[31:0]
  g0[3] = (unsigned)((ga >> 32) & 0x01ffffffu) | (2u << 30);  // addr[56:32], type=2
  i32x8 g1;
  g1[0] = (int)((1u << 16) | (1u << 20) |
                (pad_int_code << 22) | (pad_amt_code << 25)); // data_size=1, pad_enable
  g1[1] = (int)((tensor_d0 & 0xffffu) << 16);           // tensor_dim0[15:0]
  g1[2] = (int)((tensor_d0 >> 16) | ((tensor_d1 & 0xffffu) << 16));
  g1[3] = (int)((tensor_d1 >> 16) | ((tile_d0 & 0xffffu) << 16));
  g1[4] = (int)(tile_d1 & 0xffffu);                     // tile_dim1, tile_dim2=0
  g1[5] = (int)(unsigned)(stride_d0 & 0xffffffffu);     // tensor_dim0_stride lo
  g1[6] = (int)(unsigned)((stride_d0 >> 32) & 0xffffu); // stride hi, dim1_stride=0
  g1[7] = 0;
  i32x4 z4 = {0, 0, 0, 0};
  i32x8 z8 = {0, 0, 0, 0, 0, 0, 0, 0};
  __builtin_amdgcn_tensor_load_to_lds(g0, g1, z4, z4, z8, 0);
}

__device__ __forceinline__ unsigned lds_off32(const void* p) {
  // ISA §10.2: flat address low 32 bits map to the wave-relative LDS offset
  return (unsigned)(unsigned long long)p;
}

// ---------- small prep kernels ----------

__global__ void cvt_h(const float* __restrict__ s, _Float16* __restrict__ d, int n) {
  int i = blockIdx.x * blockDim.x + threadIdx.x;
  if (i < n) d[i] = (_Float16)s[i];
}

// W [K,N] f32 -> Wt [N,K] f16
__global__ void wtrans(const float* __restrict__ W, _Float16* __restrict__ Wt,
                       int K, int N) {
  int i = blockIdx.x * blockDim.x + threadIdx.x;
  if (i >= K * N) return;
  int n = i % N, k = i / N;
  Wt[(size_t)n * K + k] = (_Float16)W[i];
}

// V [B,N1,H] f32 -> Vt [B,H,N1] f16 (writes coalesced)
__global__ void vtrans(const float* __restrict__ V, _Float16* __restrict__ Vt) {
  int i = blockIdx.x * blockDim.x + threadIdx.x;   // over B*H*N1
  if (i >= BB * HHH * NN1) return;
  int n = i % NN1; int t = i / NN1; int h = t % HHH; int b = t / HHH;
  Vt[i] = (_Float16)V[((size_t)b * NN1 + n) * HHH + h];
}

// ---------- WMMA GEMM: Y(f16) = act(X(f16)[M,K] @ W[K,N] + bias), Wt is [N,K] ----------

__global__ __launch_bounds__(256) void gemm16(const _Float16* __restrict__ X,
                                              const _Float16* __restrict__ Wt,
                                              const float* __restrict__ bias,
                                              _Float16* __restrict__ Y,
                                              int M, int K, int N, int relu) {
  int wid = blockIdx.x * 8 + (threadIdx.x >> 5);
  int ntn = N >> 4;
  int row0 = (wid / ntn) << 4;
  int col0 = (wid % ntn) << 4;
  if (row0 >= M) return;
  int lane = threadIdx.x & 31, l15 = lane & 15, lh = lane >> 4;

  v8f c = {};
  for (int kk = 0; kk < K; kk += 32) {
    v16h a = load_a_tile(X, K, row0, kk);
    v16h b = load_b_tile(Wt, K, col0, kk);
    c = __builtin_amdgcn_wmma_f32_16x16x32_f16(false, a, false, b,
                                               (short)0, c, false, false);
  }
  float bn = bias[col0 + l15];
#pragma unroll
  for (int r = 0; r < 8; ++r) {
    float v = c[r] + bn;
    if (relu) v = fmaxf(v, 0.0f);
    Y[(size_t)(row0 + r + 8 * lh) * N + col0 + l15] = (_Float16)v;
  }
}

// ---------- flash attention with TDM-staged K/V tiles ----------
// 8 waves/block, 128 q rows/block. Per 32-key tile: wave0 DMAs K (16KB) and V
// (16KB) into padded LDS buffers (double-buffered, TENSORcnt-gated); each wave
// computes S = Q K^T (16 WMMA), online softmax, P restage via LDS, O += P V
// (16 WMMA) reading B-tiles from LDS.

#define KROW 264   // 512B row + 16B pad, in f16 elements
#define VROW 40    // 64B row + 16B pad, in f16 elements

__global__ __launch_bounds__(256) void flash_attn(const _Float16* __restrict__ Q,
                                                  const _Float16* __restrict__ Km,
                                                  const _Float16* __restrict__ Vt,
                                                  float* __restrict__ Out) {
  __shared__ __align__(16) _Float16 sK[2][32][KROW];    // 33792 B
  __shared__ __align__(16) _Float16 sV[2][256][VROW];   // 40960 B
  __shared__ __align__(32) _Float16 plds[8][16][32];    //  8192 B

  int w = threadIdx.x >> 5;
  int lane = threadIdx.x & 31, l15 = lane & 15, lh = lane >> 4;
  int b  = blockIdx.x >> 5;                 // 32 blocks per batch
  int q0 = (blockIdx.x & 31) * 128 + w * 16;

  const _Float16* Qb = Q  + (size_t)b * NN2 * HHH;
  const _Float16* Kb = Km + (size_t)b * NN1 * HHH;
  const _Float16* Vb = Vt + (size_t)b * HHH * NN1;

  v16h qa[8];
#pragma unroll
  for (int c = 0; c < 8; ++c) qa[c] = load_a_tile(Qb, HHH, q0, c * 32);

  v8f zero = {};
  v8f o[16];
#pragma unroll
  for (int j = 0; j < 16; ++j) o[j] = zero;
  float mrow[8], lrow[8];
#pragma unroll
  for (int r = 0; r < 8; ++r) { mrow[r] = -1e30f; lrow[r] = 0.0f; }

  const float scl = 0.0625f;                      // 1/sqrt(256)
  const float L2E = 1.44269504088896340736f;
  const int NT = NN1 / 32;                        // 128 key tiles

  // issue K+V TDM pair for tile t into buffer buf (wave-uniform args)
  auto issue_tile = [&](int t, int buf) {
    int kt = t * 32;
    // K: 32 contiguous rows x 256 f16 = one 1D 8192-elem run; pad 16B per 512B
    tdm_load_2d(lds_off32(&sK[buf][0][0]), Kb + (size_t)kt * HHH,
                /*tensor_d0=*/1u << 24, /*tensor_d1=*/1,
                /*tile_d0=*/32 * HHH,  /*tile_d1=*/1,
                /*stride_d0=*/32 * HHH,
                /*pad_int 128 DW=*/6, /*pad 4 DW=*/3);
    // V^T tile: 256 rows x 32 f16, row stride 4096 elems; pad 16B per 64B row
    tdm_load_2d(lds_off32(&sV[buf][0][0]), Vb + kt,
                /*tensor_d0=*/NN1, /*tensor_d1=*/HHH,
                /*tile_d0=*/32,    /*tile_d1=*/HHH,
                /*stride_d0=*/NN1,
                /*pad_int 16 DW=*/3, /*pad 4 DW=*/3);
  };

  if (w == 0) { issue_tile(0, 0); issue_tile(1, 1); }

  for (int it = 0; it < NT; ++it) {
    int cur = it & 1;
    if (w == 0) {
      if (it == NT - 1) __builtin_amdgcn_s_wait_tensorcnt(0);
      else              __builtin_amdgcn_s_wait_tensorcnt(2);
    }
    __syncthreads();                         // K/V tile `cur` visible to all waves

    // ---- S = Q K^T for 32 keys (B-tiles from LDS, 32B per lane, 16B aligned)
    v8f s0 = zero, s1 = zero;
#pragma unroll
    for (int c = 0; c < 8; ++c) {
      const _Float16* p0 = &sK[cur][l15][c * 32 + lh * 16];
      const _Float16* p1 = &sK[cur][16 + l15][c * 32 + lh * 16];
      v16h b0 = *(const v16h*)p0;
      v16h b1 = *(const v16h*)p1;
      s0 = __builtin_amdgcn_wmma_f32_16x16x32_f16(false, qa[c], false, b0,
                                                  (short)0, s0, false, false);
      s1 = __builtin_amdgcn_wmma_f32_16x16x32_f16(false, qa[c], false, b1,
                                                  (short)0, s1, false, false);
    }

    // ---- online softmax per row (row r+8*lh lives across the 16-lane half)
    float alpha[8];
#pragma unroll
    for (int r = 0; r < 8; ++r) {
      float z0 = s0[r] * scl, z1 = s1[r] * scl;
      float rm = fmaxf(z0, z1);
#pragma unroll
      for (int m = 1; m < 16; m <<= 1) rm = fmaxf(rm, __shfl_xor(rm, m, 32));
      float mn = fmaxf(mrow[r], rm);
      float al = exp2f((mrow[r] - mn) * L2E);
      float p0 = exp2f((z0 - mn) * L2E);
      float p1 = exp2f((z1 - mn) * L2E);
      float rs = p0 + p1;
#pragma unroll
      for (int m = 1; m < 16; m <<= 1) rs += __shfl_xor(rs, m, 32);
      lrow[r] = lrow[r] * al + rs;
      mrow[r] = mn;
      alpha[r] = al;
      plds[w][r + 8 * lh][l15]      = (_Float16)p0;   // stage P (D-layout)
      plds[w][r + 8 * lh][16 + l15] = (_Float16)p1;
    }
#pragma unroll
    for (int j = 0; j < 16; ++j)
#pragma unroll
      for (int r = 0; r < 8; ++r) o[j][r] *= alpha[r];

    // DS ops are in-order within a wave; fence compiler + wait LDS writes
    asm volatile("s_wait_dscnt 0" ::: "memory");

    // reload P in A-layout (16x32)
    const _Float16* pp = &plds[w][l15][16 * lh];
    v8h plo = *(const v8h*)pp;
    v8h phi = *(const v8h*)(pp + 8);
    v16h pa = __builtin_shufflevector(plo, phi,
                0,1,2,3,4,5,6,7,8,9,10,11,12,13,14,15);

    // ---- O += P V  (16 column tiles of d_v=256; B-tiles from padded LDS)
#pragma unroll
    for (int j = 0; j < 16; ++j) {
      const _Float16* vp = &sV[cur][j * 16 + l15][lh * 16];
      v16h vb = *(const v16h*)vp;
      o[j] = __builtin_amdgcn_wmma_f32_16x16x32_f16(false, pa, false, vb,
                                                    (short)0, o[j], false, false);
    }

    __syncthreads();                         // all waves done reading buffer `cur`
    if (w == 0 && it + 2 < NT) issue_tile(it + 2, cur);
  }

  // ---- finalize: O / l, write fp32
#pragma unroll
  for (int j = 0; j < 16; ++j)
#pragma unroll
    for (int r = 0; r < 8; ++r) {
      float v = o[j][r] / lrow[r];
      Out[((size_t)b * NN2 + q0 + r + 8 * lh) * HHH + j * 16 + l15] = v;
    }
}

// ---------- host-side launch ----------

extern "C" void kernel_launch(void* const* d_in, const int* in_sizes, int n_in,
                              void* d_out, int out_size, void* d_ws, size_t ws_size,
                              hipStream_t stream) {
  const float* x1   = (const float*)d_in[0];
  const float* x2   = (const float*)d_in[1];
  const float* v    = (const float*)d_in[2];
  const float* k_w1 = (const float*)d_in[3];
  const float* k_b1 = (const float*)d_in[4];
  const float* k_w2 = (const float*)d_in[5];
  const float* k_b2 = (const float*)d_in[6];
  const float* k_w3 = (const float*)d_in[7];
  const float* k_b3 = (const float*)d_in[8];
  const float* q_w1 = (const float*)d_in[9];
  const float* q_b1 = (const float*)d_in[10];
  const float* q_w2 = (const float*)d_in[11];
  const float* q_b2 = (const float*)d_in[12];
  const float* q_w3 = (const float*)d_in[13];
  const float* q_b3 = (const float*)d_in[14];
  float* out = (float*)d_out;

  size_t off = 0;
  auto carve = [&](size_t bytes) {
    char* p = (char*)d_ws + off;
    off += (bytes + 255) & ~(size_t)255;
    return (_Float16*)p;
  };
  _Float16* x1h = carve((size_t)MROWS * DXX * 2);
  _Float16* x2h = carve((size_t)MROWS * DXX * 2);
  _Float16* vth = carve((size_t)BB * HHH * NN1 * 2);
  _Float16* kh  = carve((size_t)MROWS * HHH * 2);
  _Float16* qh  = carve((size_t)MROWS * HHH * 2);
  _Float16* ha  = carve((size_t)MROWS * HHH * 2);
  _Float16* hb  = carve((size_t)MROWS * HHH * 2);
  _Float16* kw1t = carve((size_t)DXX * HHH * 2);
  _Float16* kw2t = carve((size_t)HHH * HHH * 2);
  _Float16* kw3t = carve((size_t)HHH * HHH * 2);
  _Float16* qw1t = carve((size_t)DXX * HHH * 2);
  _Float16* qw2t = carve((size_t)HHH * HHH * 2);
  _Float16* qw3t = carve((size_t)HHH * HHH * 2);

  const int TPB = 256;
  int nx = MROWS * DXX;
  cvt_h<<<(nx + TPB - 1) / TPB, TPB, 0, stream>>>(x1, x1h, nx);
  cvt_h<<<(nx + TPB - 1) / TPB, TPB, 0, stream>>>(x2, x2h, nx);
  int nv = BB * HHH * NN1;
  vtrans<<<(nv + TPB - 1) / TPB, TPB, 0, stream>>>(v, vth);

  int nw1 = DXX * HHH, nw = HHH * HHH;
  wtrans<<<(nw1 + TPB - 1) / TPB, TPB, 0, stream>>>(k_w1, kw1t, DXX, HHH);
  wtrans<<<(nw  + TPB - 1) / TPB, TPB, 0, stream>>>(k_w2, kw2t, HHH, HHH);
  wtrans<<<(nw  + TPB - 1) / TPB, TPB, 0, stream>>>(k_w3, kw3t, HHH, HHH);
  wtrans<<<(nw1 + TPB - 1) / TPB, TPB, 0, stream>>>(q_w1, qw1t, DXX, HHH);
  wtrans<<<(nw  + TPB - 1) / TPB, TPB, 0, stream>>>(q_w2, qw2t, HHH, HHH);
  wtrans<<<(nw  + TPB - 1) / TPB, TPB, 0, stream>>>(q_w3, qw3t, HHH, HHH);

  int gemm_blocks = (MROWS / 16) * (HHH / 16) / 8;
  gemm16<<<gemm_blocks, TPB, 0, stream>>>(x1h, kw1t, k_b1, ha, MROWS, DXX, HHH, 1);
  gemm16<<<gemm_blocks, TPB, 0, stream>>>(ha,  kw2t, k_b2, hb, MROWS, HHH, HHH, 1);
  gemm16<<<gemm_blocks, TPB, 0, stream>>>(hb,  kw3t, k_b3, kh, MROWS, HHH, HHH, 0);
  gemm16<<<gemm_blocks, TPB, 0, stream>>>(x2h, qw1t, q_b1, ha, MROWS, DXX, HHH, 1);
  gemm16<<<gemm_blocks, TPB, 0, stream>>>(ha,  qw2t, q_b2, hb, MROWS, HHH, HHH, 1);
  gemm16<<<gemm_blocks, TPB, 0, stream>>>(hb,  qw3t, q_b3, qh, MROWS, HHH, HHH, 0);

  flash_attn<<<BB * (NN2 / 128), TPB, 0, stream>>>(qh, kh, vth, out);
}
